// Mamba2EncoderBlock_9182640079497
// MI455X (gfx1250) — compile-verified, matrix operand order checked by reading, not run on hardware
//
#include <hip/hip_runtime.h>
#include <hip/hip_bf16.h>
#include <stdint.h>

#define BATCH   2
#define SEQLEN  4096
#define DMODEL  1024
#define DINNER  2048
#define NHEADS  32
#define HEADDIM 64
#define DSTATE  64
#define CONVDIM 2176
#define DPROJ   4256
#define CHUNKL  64
#define NCHUNK  (SEQLEN / CHUNKL)   // 64
#define DCONV   4
#define NROWS   (BATCH * SEQLEN)    // 8192

typedef __attribute__((ext_vector_type(16))) __bf16 v16bf;
typedef __attribute__((ext_vector_type(8)))  __bf16 v8bf;
typedef __attribute__((ext_vector_type(8)))  float  v8f;

// ---------- bf16 bit helpers (RNE) ----------
__device__ __forceinline__ unsigned short f2bfu(float f) {
  unsigned int x = __float_as_uint(f);
  x += 0x7FFFu + ((x >> 16) & 1u);
  return (unsigned short)(x >> 16);
}
__device__ __forceinline__ float bfu2f(unsigned short u) {
  return __uint_as_float(((unsigned int)u) << 16);
}
__device__ __forceinline__ __bf16 u2bf(unsigned short u) {
  union { unsigned short s; __bf16 b; } c; c.s = u; return c.b;
}

// K-index permutation inside a 16x32 bf16 WMMA operand fragment (ISA 7.12.2):
// element e (0..15) of lane (hi = lane>>4) holds K = frag_k(e,hi) within the 32-block.
__device__ __forceinline__ int frag_k(int e, int hi) {
  return (e < 8) ? (e + 8 * hi) : (e + 8 + 8 * hi);
}

// Load one 16x32 operand fragment given the lane's element pointer
// (p = row_ptr + 8*hi): two contiguous 16B loads per ISA layout.
__device__ __forceinline__ v16bf load_frag_ptr(const unsigned short* __restrict__ p) {
  v8bf lo = *reinterpret_cast<const v8bf*>(p);
  v8bf hh = *reinterpret_cast<const v8bf*>(p + 16);
  v16bf out;
#pragma unroll
  for (int i = 0; i < 8; ++i) { out[i] = lo[i]; out[8 + i] = hh[i]; }
  return out;
}

// Indexed form used by the SSD kernels (src row-major, rows=operand's non-K dim).
__device__ __forceinline__ v16bf load_frag(const unsigned short* __restrict__ src,
                                           int ld, int rbase, int k0, int lane) {
  int r  = rbase + (lane & 15);
  int hi = lane >> 4;
  return load_frag_ptr(src + (size_t)r * ld + k0 + 8 * hi);
}

__device__ __forceinline__ v8f wmma_bf16(v16bf a, v16bf b, v8f c) {
  return __builtin_amdgcn_wmma_f32_16x16x32_bf16(false, a, false, b, (short)0, c,
                                                 false, false);
}

// ---------- CDNA5 async global->LDS copy (ASYNCcnt tracked, no VGPR round-trip) ----
__device__ __forceinline__ void async_copy_b128(unsigned lds_off, const void* gaddr) {
  asm volatile("global_load_async_to_lds_b128 %0, %1, off"
               :: "v"(lds_off), "v"(gaddr) : "memory");
}
__device__ __forceinline__ void wait_async0() {
  asm volatile("s_wait_asynccnt 0" ::: "memory");
}

__device__ __forceinline__ float block_reduce(float v, float* sbuf) {
  int tid = threadIdx.x;
  sbuf[tid] = v;
  __syncthreads();
  for (int s = 128; s > 0; s >>= 1) {
    if (tid < s) sbuf[tid] += sbuf[tid + s];
    __syncthreads();
  }
  float r = sbuf[0];
  __syncthreads();
  return r;
}

// ---------- K1: LayerNorm -> bf16 ----------
__global__ __launch_bounds__(256) void ln_kernel(const float* __restrict__ x,
                                                 const float* __restrict__ w,
                                                 const float* __restrict__ b,
                                                 unsigned short* __restrict__ xn) {
  __shared__ float sbuf[256];
  int row = blockIdx.x;
  const float* xr = x + (size_t)row * DMODEL;
  float v[4];
  float s = 0.f;
#pragma unroll
  for (int i = 0; i < 4; ++i) { v[i] = xr[threadIdx.x + 256 * i]; s += v[i]; }
  float mu = block_reduce(s, sbuf) / (float)DMODEL;
  float sq = 0.f;
#pragma unroll
  for (int i = 0; i < 4; ++i) { float d = v[i] - mu; sq += d * d; }
  float var = block_reduce(sq, sbuf) / (float)DMODEL;
  float inv = rsqrtf(var + 1e-5f);
#pragma unroll
  for (int i = 0; i < 4; ++i) {
    int d = threadIdx.x + 256 * i;
    xn[(size_t)row * DMODEL + d] = f2bfu((v[i] - mu) * inv * w[d] + b[d]);
  }
}

// ---------- K2: f32 -> bf16 cast ----------
__global__ void cast_f32_bf16(const float* __restrict__ in,
                              unsigned short* __restrict__ o, int n) {
  int i = blockIdx.x * blockDim.x + threadIdx.x;
  if (i < n) o[i] = f2bfu(in[i]);
}

// ---------- K3/K11: generic WMMA GEMM  C[M,N] = A[M,K] * Bw[N,K]^T (+resid) ----------
template <int RESID>
__global__ __launch_bounds__(256) void gemm_bf16_nt(
    const unsigned short* __restrict__ A, const unsigned short* __restrict__ Bw,
    const float* __restrict__ resid, float* __restrict__ C, int M, int N, int K) {
  int lane = threadIdx.x & 31;
  int wave = threadIdx.x >> 5;
  int m0 = (blockIdx.y * 8 + wave) * 64;   // wave tile: 64 x 32
  int n0 = blockIdx.x * 32;
  int rr = lane & 15, hi = lane >> 4;

  const unsigned short* ap[4];
  const unsigned short* bp[2];
#pragma unroll
  for (int i = 0; i < 4; ++i) ap[i] = A + (size_t)(m0 + 16 * i + rr) * K + 8 * hi;
#pragma unroll
  for (int j = 0; j < 2; ++j) bp[j] = Bw + (size_t)(n0 + 16 * j + rr) * K + 8 * hi;

  v8f acc[4][2] = {};
  for (int k0 = 0; k0 < K; k0 += 32) {
    v16bf af[4], bfrag[2];
#pragma unroll
    for (int i = 0; i < 4; ++i) {
      af[i] = load_frag_ptr(ap[i]);
      __builtin_prefetch(ap[i] + 64, 0, 1);   // global_prefetch_b8: K tile two ahead (k0+64)
      ap[i] += 32;
    }
#pragma unroll
    for (int j = 0; j < 2; ++j) {
      bfrag[j] = load_frag_ptr(bp[j]);
      __builtin_prefetch(bp[j] + 64, 0, 1);
      bp[j] += 32;
    }
#pragma unroll
    for (int i = 0; i < 4; ++i)
#pragma unroll
      for (int j = 0; j < 2; ++j) acc[i][j] = wmma_bf16(af[i], bfrag[j], acc[i][j]);
  }

  int nn = lane & 15;
#pragma unroll
  for (int i = 0; i < 4; ++i) {
    size_t rowoff = (size_t)(m0 + 16 * i + 8 * hi) * N + (n0 + nn);
    float* crow = C + rowoff;
    const float* rrow = resid + rowoff;
#pragma unroll
    for (int r8 = 0; r8 < 8; ++r8) {
#pragma unroll
      for (int j = 0; j < 2; ++j) {
        float v = acc[i][j][r8];
        if (RESID) v += rrow[16 * j];
        crow[16 * j] = v;
      }
      crow += N;
      if (RESID) rrow += N;
    }
  }
}

// ---------- K4: dt = softplus(raw + bias); dtA = dt * (-exp(A_log)) ----------
__global__ void dt_kernel(const float* __restrict__ zx, const float* __restrict__ dt_bias,
                          const float* __restrict__ A_log, float* __restrict__ dtv,
                          float* __restrict__ dtA) {
  int i = blockIdx.x * 256 + threadIdx.x;   // NROWS * NHEADS
  int row = i >> 5, h = i & 31;
  float rawv = zx[(size_t)row * DPROJ + (DINNER + CONVDIM) + h] + dt_bias[h];
  float d = (rawv > 20.f) ? rawv : log1pf(__expf(rawv));
  dtv[i] = d;
  dtA[i] = d * (-__expf(A_log[h]));
}

// ---------- K5: per-chunk inclusive cumsum of dtA ----------
__global__ void cumsum_kernel(const float* __restrict__ dtA, float* __restrict__ Acum) {
  int i = blockIdx.x * 256 + threadIdx.x;   // BATCH*NHEADS*NCHUNK = 4096
  if (i >= BATCH * NHEADS * NCHUNK) return;
  int b = i / (NHEADS * NCHUNK);
  int h = (i / NCHUNK) % NHEADS;
  int c = i % NCHUNK;
  float run = 0.f;
  float* o = Acum + (size_t)i * CHUNKL;     // layout [b,h,c,t]
  for (int t = 0; t < CHUNKL; ++t) {
    run += dtA[((size_t)(b * SEQLEN + c * CHUNKL + t)) * NHEADS + h];
    o[t] = run;
  }
}

// ---------- K6: causal depthwise conv + SiLU, scatter to xs/xdt/Bq/Cq ----------
__global__ void conv_kernel(const float* __restrict__ zx, const float* __restrict__ cw,
                            const float* __restrict__ cb, const float* __restrict__ dtv,
                            float* __restrict__ xs, unsigned short* __restrict__ xdt,
                            unsigned short* __restrict__ Bq, unsigned short* __restrict__ Cq) {
  size_t i = (size_t)blockIdx.x * 256 + threadIdx.x;
  if (i >= (size_t)NROWS * CONVDIM) return;
  int row = (int)(i / CONVDIM);
  int ch  = (int)(i % CONVDIM);
  int l = row & (SEQLEN - 1);
  float acc = cb[ch];
#pragma unroll
  for (int j = 0; j < DCONV; ++j) {
    int li = l - (DCONV - 1) + j;
    if (li >= 0)
      acc += zx[(size_t)(row - (DCONV - 1) + j) * DPROJ + DINNER + ch] * cw[ch * DCONV + j];
  }
  float v = acc / (1.f + __expf(-acc));   // SiLU
  if (ch < DINNER) {
    xs[(size_t)row * DINNER + ch] = v;
    int h = ch >> 6;
    xdt[(size_t)row * DINNER + ch] = f2bfu(v * dtv[row * NHEADS + h]);
  } else if (ch < DINNER + DSTATE) {
    Bq[(size_t)row * DSTATE + (ch - DINNER)] = f2bfu(v);
  } else {
    Cq[(size_t)row * DSTATE + (ch - DINNER - DSTATE)] = f2bfu(v);
  }
}

// ---------- K7: per-(b,c) chunk: G = Cq*Bq^T, then per head Y_diag & states ----------
__global__ __launch_bounds__(256) void chunk_kernel(
    const unsigned short* __restrict__ Cq, const unsigned short* __restrict__ Bq,
    const unsigned short* __restrict__ xdt, const float* __restrict__ Acum,
    float* __restrict__ ydiag, float* __restrict__ states) {
  __shared__ __align__(16) unsigned short sCq[CHUNKL * DSTATE];
  __shared__ __align__(16) unsigned short sBq[CHUNKL * DSTATE];
  __shared__ float sG[CHUNKL * CHUNKL];
  __shared__ float sAc[8][CHUNKL];
  int lane = threadIdx.x & 31;
  int wave = threadIdx.x >> 5;
  int b = blockIdx.x / NCHUNK;
  int c = blockIdx.x % NCHUNK;
  int rbase = b * SEQLEN + c * CHUNKL;

  { // stage Cq/Bq chunk tiles via async global->LDS (ASYNCcnt), 16B per lane per op
    unsigned sC_off = (unsigned)(uintptr_t)(void*)sCq;
    unsigned sB_off = (unsigned)(uintptr_t)(void*)sBq;
    const char* gC = (const char*)(Cq + (size_t)rbase * DSTATE);
    const char* gB = (const char*)(Bq + (size_t)rbase * DSTATE);
    unsigned t16 = threadIdx.x * 16;
    async_copy_b128(sC_off + t16,        gC + t16);
    async_copy_b128(sC_off + t16 + 4096, gC + t16 + 4096);
    async_copy_b128(sB_off + t16,        gB + t16);
    async_copy_b128(sB_off + t16 + 4096, gB + t16 + 4096);
    wait_async0();
  }
  __syncthreads();

  { // G[l][s] = sum_n Cq[l][n] * Bq[s][n]   (16 tiles of 16x16, 2 per wave)
    int mi  = wave >> 1;
    int ni0 = (wave & 1) * 2;
    v8f g0 = {}, g1 = {};
    for (int k0 = 0; k0 < DSTATE; k0 += 32) {
      v16bf a  = load_frag(sCq, DSTATE, 16 * mi, k0, lane);
      v16bf b0 = load_frag(sBq, DSTATE, 16 * ni0, k0, lane);
      v16bf b1 = load_frag(sBq, DSTATE, 16 * (ni0 + 1), k0, lane);
      g0 = wmma_bf16(a, b0, g0);
      g1 = wmma_bf16(a, b1, g1);
    }
    int hi = lane >> 4, nn = lane & 15;
#pragma unroll
    for (int r = 0; r < 8; ++r) {
      int l = 16 * mi + r + 8 * hi;
      sG[l * CHUNKL + 16 * ni0 + nn]       = g0[r];
      sG[l * CHUNKL + 16 * (ni0 + 1) + nn] = g1[r];
    }
  }
  __syncthreads();

  for (int hh = 0; hh < 4; ++hh) {   // each wave handles 4 heads
    int h = wave * 4 + hh;
    const float* ac = Acum + ((size_t)(b * NHEADS + h) * NCHUNK + c) * CHUNKL;
    sAc[wave][lane]      = ac[lane];
    sAc[wave][lane + 32] = ac[lane + 32];
    __syncthreads();
    int hi = lane >> 4, nn = lane & 15;
    const unsigned short* xp = xdt + ((size_t)rbase * NHEADS + h) * HEADDIM;

    // ---- Y_diag = (G .* exp(segsum) masked) @ Xdt ----
    v8f ya[4][4] = {};
    for (int k0 = 0; k0 < CHUNKL; k0 += 32) {
      v16bf af[4];
#pragma unroll
      for (int mi = 0; mi < 4; ++mi) {
        int l = 16 * mi + nn;
        float acl = sAc[wave][l];
        v16bf f;
#pragma unroll
        for (int e = 0; e < 16; ++e) {
          int s = k0 + frag_k(e, hi);
          float v = (s <= l) ? __expf(acl - sAc[wave][s]) * sG[l * CHUNKL + s] : 0.f;
          f[e] = u2bf(f2bfu(v));
        }
        af[mi] = f;
      }
      v16bf bx[4];
#pragma unroll
      for (int nj = 0; nj < 4; ++nj) {
        int p = 16 * nj + nn;
        v16bf f;
#pragma unroll
        for (int e = 0; e < 16; ++e) {
          int s = k0 + frag_k(e, hi);
          f[e] = u2bf(xp[(size_t)s * (NHEADS * HEADDIM) + p]);
        }
        bx[nj] = f;
      }
#pragma unroll
      for (int mi = 0; mi < 4; ++mi)
#pragma unroll
        for (int nj = 0; nj < 4; ++nj) ya[mi][nj] = wmma_bf16(af[mi], bx[nj], ya[mi][nj]);
    }
#pragma unroll
    for (int mi = 0; mi < 4; ++mi)
#pragma unroll
      for (int nj = 0; nj < 4; ++nj)
#pragma unroll
        for (int r = 0; r < 8; ++r) {
          int l = 16 * mi + r + 8 * hi;
          int p = 16 * nj + nn;
          ydiag[((size_t)(rbase + l) * NHEADS + h) * HEADDIM + p] = ya[mi][nj][r];
        }

    // ---- states[n][p] = sum_s Bq[s][n] * exp(Acum[63]-Acum[s]) * Xdt[s][p] ----
    float ac63 = sAc[wave][63];
    v8f sacc[4][4] = {};
    for (int k0 = 0; k0 < CHUNKL; k0 += 32) {
      v16bf af[4];
#pragma unroll
      for (int mi = 0; mi < 4; ++mi) {
        int n = 16 * mi + nn;
        v16bf f;
#pragma unroll
        for (int e = 0; e < 16; ++e) {
          int s = k0 + frag_k(e, hi);
          float v = bfu2f(sBq[s * DSTATE + n]) * __expf(ac63 - sAc[wave][s]);
          f[e] = u2bf(f2bfu(v));
        }
        af[mi] = f;
      }
      v16bf bx[4];
#pragma unroll
      for (int nj = 0; nj < 4; ++nj) {
        int p = 16 * nj + nn;
        v16bf f;
#pragma unroll
        for (int e = 0; e < 16; ++e) {
          int s = k0 + frag_k(e, hi);
          f[e] = u2bf(xp[(size_t)s * (NHEADS * HEADDIM) + p]);
        }
        bx[nj] = f;
      }
#pragma unroll
      for (int mi = 0; mi < 4; ++mi)
#pragma unroll
        for (int nj = 0; nj < 4; ++nj) sacc[mi][nj] = wmma_bf16(af[mi], bx[nj], sacc[mi][nj]);
    }
    float* sdst = states + ((size_t)(b * NCHUNK + c) * NHEADS + h) * (DSTATE * HEADDIM);
#pragma unroll
    for (int mi = 0; mi < 4; ++mi)
#pragma unroll
      for (int nj = 0; nj < 4; ++nj)
#pragma unroll
        for (int r = 0; r < 8; ++r) {
          int n = 16 * mi + r + 8 * hi;
          int p = 16 * nj + nn;
          sdst[(size_t)n * HEADDIM + p] = sacc[mi][nj][r];
        }
    __syncthreads();
  }
}

// ---------- K8: sequential inter-chunk scan; emit transposed prev-states (bf16) ----------
__global__ void scan_kernel(const float* __restrict__ states, const float* __restrict__ Acum,
                            unsigned short* __restrict__ prevT) {
  int i = blockIdx.x * 256 + threadIdx.x;   // BATCH*NHEADS*DSTATE*HEADDIM
  int p = i & 63;
  int n = (i >> 6) & 63;
  int h = (i >> 12) & 31;
  int b = i >> 17;
  float acc = 0.f;
  for (int c = 0; c < NCHUNK; ++c) {
    prevT[(((size_t)(b * NCHUNK + c) * NHEADS + h) * HEADDIM + p) * DSTATE + n] = f2bfu(acc);
    float dec = __expf(Acum[((size_t)(b * NHEADS + h) * NCHUNK + c) * CHUNKL + 63]);
    acc = dec * acc + states[(((size_t)(b * NCHUNK + c) * NHEADS + h) * DSTATE + n) * HEADDIM + p];
  }
}

// ---------- K9: Y_off = exp(Acum) * (Cq @ prevS) ; y = Ydiag + Y_off + D*xs ----------
__global__ __launch_bounds__(256) void yoff_kernel(
    const unsigned short* __restrict__ Cq, const unsigned short* __restrict__ prevT,
    const float* __restrict__ Acum, const float* __restrict__ xs,
    const float* __restrict__ D_skip, float* __restrict__ y) {
  int lane = threadIdx.x & 31;
  int wave = threadIdx.x >> 5;
  int task = blockIdx.x * 8 + wave;        // (b*NCHUNK + c)*NHEADS + h
  int h = task & 31;
  int c = (task >> 5) & 63;
  int b = task >> 11;
  int rbase = b * SEQLEN + c * CHUNKL;
  const unsigned short* As = Cq + (size_t)rbase * DSTATE;                 // [l][n]
  const unsigned short* Bs = prevT + (size_t)task * (HEADDIM * DSTATE);   // [p][n] = prevS^T
  v8f acc[4][4] = {};
  for (int k0 = 0; k0 < DSTATE; k0 += 32) {
    v16bf af[4], bfrag[4];
#pragma unroll
    for (int mi = 0; mi < 4; ++mi) af[mi] = load_frag(As, DSTATE, 16 * mi, k0, lane);
#pragma unroll
    for (int nj = 0; nj < 4; ++nj) bfrag[nj] = load_frag(Bs, DSTATE, 16 * nj, k0, lane);
#pragma unroll
    for (int mi = 0; mi < 4; ++mi)
#pragma unroll
      for (int nj = 0; nj < 4; ++nj) acc[mi][nj] = wmma_bf16(af[mi], bfrag[nj], acc[mi][nj]);
  }
  int hi = lane >> 4, nn = lane & 15;
  const float* ac = Acum + ((size_t)(b * NHEADS + h) * NCHUNK + c) * CHUNKL;
#pragma unroll
  for (int mi = 0; mi < 4; ++mi)
#pragma unroll
    for (int r = 0; r < 8; ++r) {
      int ll = 16 * mi + r + 8 * hi;
      float ea = __expf(ac[ll]);
#pragma unroll
      for (int nj = 0; nj < 4; ++nj) {
        int p = 16 * nj + nn;
        size_t idx = (size_t)(rbase + ll) * DINNER + (h * HEADDIM + p);
        y[idx] = ea * acc[mi][nj][r] + y[idx] + D_skip[h] * xs[idx];
      }
    }
}

// ---------- K10: gate with SiLU(z), RMSNorm, cast to bf16 ----------
__global__ __launch_bounds__(256) void gate_rms_kernel(
    const float* __restrict__ y, const float* __restrict__ zx,
    const float* __restrict__ norm_w, unsigned short* __restrict__ ygb) {
  __shared__ float sbuf[256];
  int row = blockIdx.x;
  float yv[8];
  float sq = 0.f;
#pragma unroll
  for (int i = 0; i < 8; ++i) {
    int d = threadIdx.x + 256 * i;
    float z = zx[(size_t)row * DPROJ + d];
    float g = y[(size_t)row * DINNER + d] * (z / (1.f + __expf(-z)));
    yv[i] = g;
    sq += g * g;
  }
  float tot = block_reduce(sq, sbuf);
  float scale = rsqrtf(tot / (float)DINNER + 1e-5f);
#pragma unroll
  for (int i = 0; i < 8; ++i) {
    int d = threadIdx.x + 256 * i;
    ygb[(size_t)row * DINNER + d] = f2bfu(yv[i] * scale * norm_w[d]);
  }
}

extern "C" void kernel_launch(void* const* d_in, const int* in_sizes, int n_in,
                              void* d_out, int out_size, void* d_ws, size_t ws_size,
                              hipStream_t stream) {
  (void)in_sizes; (void)n_in; (void)out_size; (void)ws_size;
  const float* x          = (const float*)d_in[0];
  const float* ln_w       = (const float*)d_in[1];
  const float* ln_b       = (const float*)d_in[2];
  const float* in_proj_w  = (const float*)d_in[3];
  const float* conv_w     = (const float*)d_in[4];
  const float* conv_b     = (const float*)d_in[5];
  const float* dt_bias    = (const float*)d_in[6];
  const float* A_log      = (const float*)d_in[7];
  const float* D_skip     = (const float*)d_in[8];
  const float* norm_w     = (const float*)d_in[9];
  const float* out_proj_w = (const float*)d_in[10];
  float* out = (float*)d_out;

  char* base = (char*)d_ws;
  size_t off = 0;
  auto nxt = [&](size_t bytes) -> void* {
    void* r = base + off;
    off = (off + bytes + 255) & ~(size_t)255;
    return r;
  };
  unsigned short* xn_bf   = (unsigned short*)nxt((size_t)NROWS * DMODEL * 2);
  unsigned short* win_bf  = (unsigned short*)nxt((size_t)DPROJ * DMODEL * 2);
  unsigned short* wout_bf = (unsigned short*)nxt((size_t)DMODEL * DINNER * 2);
  float* zx    = (float*)nxt((size_t)NROWS * DPROJ * 4);
  float* xs    = (float*)nxt((size_t)NROWS * DINNER * 4);
  unsigned short* xdt = (unsigned short*)nxt((size_t)NROWS * DINNER * 2);
  float* dtv   = (float*)nxt((size_t)NROWS * NHEADS * 4);
  float* dtA   = (float*)nxt((size_t)NROWS * NHEADS * 4);
  unsigned short* Bqb = (unsigned short*)nxt((size_t)NROWS * DSTATE * 2);
  unsigned short* Cqb = (unsigned short*)nxt((size_t)NROWS * DSTATE * 2);
  float* Acum  = (float*)nxt((size_t)BATCH * NHEADS * NCHUNK * CHUNKL * 4);
  float* states = (float*)nxt((size_t)BATCH * NCHUNK * NHEADS * DSTATE * HEADDIM * 4);
  unsigned short* prevT = (unsigned short*)nxt((size_t)BATCH * NCHUNK * NHEADS * HEADDIM * DSTATE * 2);
  float* ydiag = (float*)nxt((size_t)NROWS * DINNER * 4);
  unsigned short* ygb = (unsigned short*)nxt((size_t)NROWS * DINNER * 2);

  ln_kernel<<<NROWS, 256, 0, stream>>>(x, ln_w, ln_b, xn_bf);
  cast_f32_bf16<<<(DPROJ * DMODEL + 255) / 256, 256, 0, stream>>>(in_proj_w, win_bf, DPROJ * DMODEL);
  cast_f32_bf16<<<(DMODEL * DINNER + 255) / 256, 256, 0, stream>>>(out_proj_w, wout_bf, DMODEL * DINNER);
  gemm_bf16_nt<0><<<dim3(DPROJ / 32, NROWS / 512), 256, 0, stream>>>(xn_bf, win_bf, zx, zx,
                                                                     NROWS, DPROJ, DMODEL);
  dt_kernel<<<(NROWS * NHEADS) / 256, 256, 0, stream>>>(zx, dt_bias, A_log, dtv, dtA);
  cumsum_kernel<<<(BATCH * NHEADS * NCHUNK + 255) / 256, 256, 0, stream>>>(dtA, Acum);
  conv_kernel<<<(int)(((size_t)NROWS * CONVDIM + 255) / 256), 256, 0, stream>>>(
      zx, conv_w, conv_b, dtv, xs, xdt, Bqb, Cqb);
  chunk_kernel<<<BATCH * NCHUNK, 256, 0, stream>>>(Cqb, Bqb, xdt, Acum, ydiag, states);
  scan_kernel<<<(BATCH * NHEADS * DSTATE * HEADDIM) / 256, 256, 0, stream>>>(states, Acum, prevT);
  yoff_kernel<<<(BATCH * NCHUNK * NHEADS) / 8, 256, 0, stream>>>(Cqb, prevT, Acum, xs, D_skip, ydiag);
  gate_rms_kernel<<<NROWS, 256, 0, stream>>>(ydiag, zx, norm_w, ygb);
  gemm_bf16_nt<1><<<dim3(DMODEL / 32, NROWS / 512), 256, 0, stream>>>(ygb, wout_bf, x, out,
                                                                      NROWS, DMODEL, DINNER);
}
